// GMMWeightedCondLarge_63745904607833
// MI455X (gfx1250) — matrix operand order repaired and checked by reading, not compiled, not run on hardware
//
#include <hip/hip_runtime.h>
#include <hip/hip_bf16.h>

// ---------------------------------------------------------------------------
// Fused GMM-conditioned sampler for MI455X (gfx1250, wave32, WMMA).
// One fused kernel: 5-layer MLP entirely in LDS with bf16 v_wmma_f32_16x16x32,
// then per-sample Box-Muller / Lambert concentric-disk sampling + logsumexp.
// Weight staging uses GLOBAL_LOAD_ASYNC_TO_LDS_B128 (ASYNCcnt) when the
// toolchain exposes the builtins, else a synchronous b128 copy.
// ---------------------------------------------------------------------------

typedef __attribute__((ext_vector_type(16))) __bf16 v16bf;
typedef __attribute__((ext_vector_type(8)))  float  v8f;
typedef int v4i __attribute__((__vector_size__(16)));

#define NTOT      262144LL
#define TILE_N    128
#define NTHREADS  256

// workspace layout in bf16 (ushort) elements: transposed weights Wt[out][in]
#define OFF0 0        // Wt0 [128][64]
#define OFF1 8192     // Wt1 [512][128]
#define OFF2 73728    // Wt2 [128][512]
#define OFF3 139264   // Wt3 [32][128]
#define OFF4 143360   // Wt4 [48][32] (rows 41..47 zero padded)
// total 144896 ushort = 289792 bytes of d_ws

#if defined(__gfx1250__) && \
    __has_builtin(__builtin_amdgcn_global_load_async_to_lds_b128) && \
    __has_builtin(__builtin_amdgcn_s_wait_asynccnt)
#define USE_ASYNC_STAGE 1
typedef __attribute__((address_space(1))) v4i as1_v4i;   // global int4
typedef __attribute__((address_space(3))) v4i as3_v4i;   // LDS int4
#else
#define USE_ASYNC_STAGE 0
#endif

__device__ __forceinline__ unsigned short f2bf(float f) {
  union { float f; unsigned int u; } v; v.f = f;
  unsigned int r = v.u + 0x7FFFu + ((v.u >> 16) & 1u);  // RNE
  return (unsigned short)(r >> 16);
}

// Fragment register image: 8 VGPRs = two 16B chunks, loaded as ds_load_b128.
union Frag { v16bf v; uint4 q[2]; };

// Stage nElems bf16 weights global->LDS (both 16B aligned).
__device__ __forceinline__ void stage_weights(const unsigned short* __restrict__ src_g,
                                              unsigned short* __restrict__ dst_l,
                                              int nElems, int tid)
{
  const uint4* src = (const uint4*)src_g;
  uint4*       dst = (uint4*)dst_l;
  const int nv = nElems / 8;
#if USE_ASYNC_STAGE
  for (int i = tid; i < nv; i += NTHREADS) {
    as1_v4i* g = (as1_v4i*)(uintptr_t)(src + i);
    as3_v4i* l = (as3_v4i*)(unsigned int)(uintptr_t)(dst + i);   // low 32b = LDS offset
    __builtin_amdgcn_global_load_async_to_lds_b128(g, l, 0, 0);
  }
  __builtin_amdgcn_s_wait_asynccnt(0);
#else
  for (int i = tid; i < nv; i += NTHREADS) dst[i] = src[i];
#endif
}

// One MLP layer: actIn [128 x CIN] bf16 (LDS) x Wt [COUT x CIN] bf16 (global,
// staged to LDS per chunk) -> actOut [128 x COUT] bf16 (LDS), or f32 fout when FINAL.
template<int CIN, int COUT, int COUTL, bool RELU, bool FINAL>
__device__ __forceinline__ void mlp_layer(
    const unsigned short* __restrict__ actIn,
    unsigned short*       __restrict__ actOut,
    float*                __restrict__ fout,
    const unsigned short* __restrict__ Wg,
    const float*          __restrict__ bias,
    unsigned short*       __restrict__ wlds,
    int tid)
{
  constexpr int CMAX0 = 32768 / CIN;                       // cols fitting 64KB LDS stage
  constexpr int CH0   = (COUT < CMAX0) ? COUT : CMAX0;
  constexpr int CHUNK = (CH0 < 128) ? CH0 : 128;           // cap accumulators at 8 tiles
  constexpr int NJ    = CHUNK / 16;
  constexpr int NK    = CIN / 32;
  constexpr int NCH   = COUT / CHUNK;

  const int lane = tid & 31;
  const int wave = tid >> 5;          // 8 waves, each owns 16 rows
  const int half = lane >> 4;
  const int m    = lane & 15;

  for (int ch = 0; ch < NCH; ++ch) {
    const int c0 = ch * CHUNK;
    stage_weights(Wg + c0 * CIN, wlds, CHUNK * CIN, tid);
    __syncthreads();

    v8f acc[NJ] = {};
    #pragma unroll
    for (int kt = 0; kt < NK; ++kt) {
      // A fragment (16x32 bf16, ISA 7.12.2 zig-zag): two b128 chunks at
      // +8*half ushorts and +8*half+16 ushorts.
      Frag a;
      const uint4* ap = (const uint4*)(actIn + (wave * 16 + m) * CIN + kt * 32 + 8 * half);
      a.q[0] = ap[0];
      a.q[1] = ap[2];

      #pragma unroll
      for (int j = 0; j < NJ; ++j) {
        // B fragment (32x16 bf16): same zig-zag as A -> b128 at +8*half and +8*half+16.
        Frag b;
        const uint4* bp = (const uint4*)(wlds + (j * 16 + m) * CIN + kt * 32 + 8 * half);
        b.q[0] = bp[0];
        b.q[1] = bp[2];

        acc[j] = __builtin_amdgcn_wmma_f32_16x16x32_bf16(
            false, a.v, false, b.v, (short)0, acc[j], false, false);
      }
    }

    // epilogue: bias (+ReLU), D layout: col = lane&15, row = r + 8*half
    #pragma unroll
    for (int j = 0; j < NJ; ++j) {
      const int gcol = c0 + j * 16 + m;
      float bv;
      if constexpr (FINAL) bv = (gcol < COUTL) ? bias[gcol] : 0.0f;
      else                 bv = bias[gcol];
      #pragma unroll
      for (int r = 0; r < 8; ++r) {
        float val = acc[j][r] + bv;
        if constexpr (RELU) val = fmaxf(val, 0.0f);
        const int row = wave * 16 + r + 8 * half;
        if constexpr (FINAL) fout[row * COUT + gcol] = val;
        else                 actOut[row * COUT + gcol] = f2bf(val);
      }
    }
    __syncthreads();
  }
}

__global__ __launch_bounds__(NTHREADS) void gmm_fused_kernel(
    const float* __restrict__ cond,
    const float* __restrict__ rs,
    const unsigned short* __restrict__ wts,
    const float* __restrict__ b0, const float* __restrict__ b1,
    const float* __restrict__ b2, const float* __restrict__ b3,
    const float* __restrict__ b4,
    float* __restrict__ out)
{
  extern __shared__ unsigned short smem[];
  unsigned short* buf0 = smem;                     // 65536 ushort = 128KB (128 x 512 max)
  unsigned short* buf1 = smem + 65536;             // 16384 ushort = 32KB  (128 x 128 max)
  unsigned short* wlds = smem + 65536 + 16384;     // 32768 ushort = 64KB weight stage
  float* fout = (float*)buf1;                      // final layer f32: 128 x 48 = 24KB

  const int tid = threadIdx.x;
  const long long nbase = (long long)blockIdx.x * TILE_N;

  // stage input tile 128 x 64 (f32 -> bf16), coalesced
  for (int i = tid; i < TILE_N * 64; i += NTHREADS) {
    const int row = i >> 6, col = i & 63;
    buf0[i] = f2bf(cond[(nbase + row) * 64 + col]);
  }
  // warm L2/L0 for the first weight stage
  for (int i = tid * 64; i < 8192; i += NTHREADS * 64)
    __builtin_prefetch(wts + OFF0 + i, 0, 1);
  __syncthreads();

  mlp_layer< 64, 128, 128, true,  false>(buf0, buf1, nullptr, wts + OFF0, b0, wlds, tid);
  mlp_layer<128, 512, 512, true,  false>(buf1, buf0, nullptr, wts + OFF1, b1, wlds, tid);
  mlp_layer<512, 128, 128, true,  false>(buf0, buf1, nullptr, wts + OFF2, b2, wlds, tid);
  mlp_layer<128,  32,  32, true,  false>(buf1, buf0, nullptr, wts + OFF3, b3, wlds, tid);
  mlp_layer< 32,  48,  41, false, true >(buf0, nullptr, fout, wts + OFF4, b4, wlds, tid);
  __syncthreads();

  // ---- per-sample sampling + logsumexp tail (128 rows, one thread each) ----
  if (tid < TILE_N) {
    const int row = tid;
    const long long n = nbase + row;
    const float* h = fout + row * 48;

    float loc0[8], loc1[8], ls0[8], ls1[8];
    #pragma unroll
    for (int k = 0; k < 8; ++k) {
      loc0[k] = h[2 * k];      loc1[k] = h[2 * k + 1];
      ls0[k]  = h[16 + 2 * k]; ls1[k]  = h[17 + 2 * k];
    }
    float wr[9], wsum = 0.0f;
    #pragma unroll
    for (int k = 0; k < 9; ++k) { wr[k] = fabsf(h[32 + k]); wsum += wr[k]; }
    const float winv = 1.0f / wsum;
    const float wL = wr[8] * winv;

    const float r0 = rs[2 * n], r1 = rs[2 * n + 1];
    const bool lam = r0 < wL;
    const float u0 = lam ? (r0 / wL) : ((r0 - wL) / (1.0f - wL));
    float U1 = lam ? 0.5f : u0;
    U1 = fmaxf(U1, 1e-12f);
    const float R  = sqrtf(-2.0f * logf(U1));
    const float th = 6.28318530717958647692f * r1;
    float sth, cth; sincosf(th, &sth, &cth);

    float ss0 = 0, ss1 = 0, lm0 = 0, lm1 = 0;
    #pragma unroll
    for (int k = 0; k < 8; ++k) {
      ss0 += expf(ls0[k]); ss1 += expf(ls1[k]);
      lm0 += loc0[k];      lm1 += loc1[k];
    }
    const float zg0 = (R * cth) * ss0 + lm0;
    const float zg1 = (R * sth) * ss1 + lm1;

    const float wo0 = u0 * 2.0f - 1.0f, wo1 = r1 * 2.0f - 1.0f;
    const bool nz = !((wo0 == 0.0f) && (wo1 == 0.0f));
    const bool c1 = (fabsf(wo0) > fabsf(wo1)) && nz;
    const bool c2 = (!c1) && nz;
    const float sf0 = (wo0 == 0.0f) ? 1.0f : wo0;
    const float sf1 = (wo1 == 0.0f) ? 1.0f : wo1;
    const float phi = c1 ? (0.78539816339f * wo1 / sf0)
                         : (1.57079632679f - 0.78539816339f * wo0 / sf1);
    const float rr = c1 ? wo0 : (c2 ? wo1 : 0.0f);
    float sph, cph; sincosf(phi, &sph, &cph);
    const float z0 = lam ? rr * cph : zg0;
    const float z1 = lam ? rr * sph : zg1;

    float lp[9];
    #pragma unroll
    for (int k = 0; k < 8; ++k) {
      const float e0 = (z0 - loc0[k]) * expf(-ls0[k]);
      const float e1 = (z1 - loc1[k]) * expf(-ls1[k]);
      lp[k] = -1.83787706640934548356f               // -0.5*d*log(2*pi), d=2
              + logf(wr[k] * winv + 1e-5f)
              - 0.5f * (e0 * e0 + e1 * e1)
              - (ls0[k] + ls1[k]);
    }
    const float pdf = (z0 * z0 + z1 * z1 > 1.0f) ? 0.0f : 0.31830988618f;
    lp[8] = logf(pdf + 1e-5f) + logf(wL);

    float mx = lp[0];
    #pragma unroll
    for (int k = 1; k < 9; ++k) mx = fmaxf(mx, lp[k]);
    float se = 0.0f;
    #pragma unroll
    for (int k = 0; k < 9; ++k) se += expf(lp[k] - mx);
    const float lse = mx + logf(se);

    out[2 * n]     = z0;
    out[2 * n + 1] = z1;
    out[2 * NTOT + n] = lse;
  }
}

// transpose + f32->bf16 convert: o[oc*In + k] = W[k*Out + oc], pad oc>=Out with 0
__global__ void prep_wt_kernel(const float* __restrict__ W,
                               unsigned short* __restrict__ o,
                               int In, int Out, int OutPad)
{
  const int i = blockIdx.x * 256 + threadIdx.x;
  if (i >= In * OutPad) return;
  const int oc = i / In, k = i - oc * In;
  const float v = (oc < Out) ? W[k * Out + oc] : 0.0f;
  o[oc * In + k] = f2bf(v);
}

extern "C" void kernel_launch(void* const* d_in, const int* in_sizes, int n_in,
                              void* d_out, int out_size, void* d_ws, size_t ws_size,
                              hipStream_t stream)
{
  (void)in_sizes; (void)n_in; (void)out_size; (void)ws_size;
  const float* cond = (const float*)d_in[0];
  const float* rs   = (const float*)d_in[1];
  const float* W0 = (const float*)d_in[2];  const float* b0 = (const float*)d_in[3];
  const float* W1 = (const float*)d_in[4];  const float* b1 = (const float*)d_in[5];
  const float* W2 = (const float*)d_in[6];  const float* b2 = (const float*)d_in[7];
  const float* W3 = (const float*)d_in[8];  const float* b3 = (const float*)d_in[9];
  const float* W4 = (const float*)d_in[10]; const float* b4 = (const float*)d_in[11];
  unsigned short* ws = (unsigned short*)d_ws;

  auto prep = [&](const float* W, unsigned short* o, int In, int Out, int OutPad) {
    const int total = In * OutPad;
    prep_wt_kernel<<<(total + 255) / 256, 256, 0, stream>>>(W, o, In, Out, OutPad);
  };
  prep(W0, ws + OFF0,  64, 128, 128);
  prep(W1, ws + OFF1, 128, 512, 512);
  prep(W2, ws + OFF2, 512, 128, 128);
  prep(W3, ws + OFF3, 128,  32,  32);
  prep(W4, ws + OFF4,  32,  41,  48);

  constexpr int SMEM_BYTES = (65536 + 16384 + 32768) * 2;  // 224KB of 320KB/WGP
  hipError_t err = hipFuncSetAttribute(reinterpret_cast<const void*>(gmm_fused_kernel),
                                       hipFuncAttributeMaxDynamicSharedMemorySize, SMEM_BYTES);
  (void)err;
  gmm_fused_kernel<<<(int)(NTOT / TILE_N), NTHREADS, SMEM_BYTES, stream>>>(
      cond, rs, ws, b0, b1, b2, b3, b4, (float*)d_out);
}
